// Attention_33328946217680
// MI455X (gfx1250) — matrix operand order repaired and checked by reading, not compiled, not run on hardware
//
#include <hip/hip_runtime.h>
#include <hip/hip_bf16.h>

typedef __attribute__((ext_vector_type(16))) _Float16 v16h;
typedef __attribute__((ext_vector_type(8)))  float    v8f;
typedef __attribute__((ext_vector_type(4)))  _Float16 h4;
typedef __attribute__((ext_vector_type(2)))  _Float16 h2;
typedef __attribute__((ext_vector_type(4)))  unsigned int v4u;
typedef __attribute__((ext_vector_type(8)))  int      v8i;
typedef __attribute__((ext_vector_type(4)))  int      v4i;

#define BATCH 2
#define NLEN  2048
#define SLEN  2048
#define NH    8
#define HD    128
#define DIM   1024
#define BQ    128   // query rows per workgroup (16 per wave * 8 waves)
#define BK    32    // keys per block iteration (one WMMA K-dim)
#define NBLK  (SLEN / BK)

#if defined(__has_builtin)
#if __has_builtin(__builtin_amdgcn_tensor_load_to_lds)
#define HAVE_TDM 1
#endif
#if __has_builtin(__builtin_amdgcn_s_wait_tensorcnt)
#define HAVE_TCNT_BUILTIN 1
#endif
#endif

#if __has_include(<hip/amd_detail/amd_gfx1250_TDM.h>)
#define TDM_6ARG 1
#endif

__device__ __forceinline__ v8f wmma_f16(v16h a, v16h b, v8f c) {
    return __builtin_amdgcn_wmma_f32_16x16x32_f16(false, a, false, b, (short)0, c, false, false);
}

__device__ __forceinline__ void wait_tensorcnt0() {
#if defined(HAVE_TCNT_BUILTIN)
    __builtin_amdgcn_s_wait_tensorcnt(0);
#else
    asm volatile("s_wait_tensorcnt 0x0" ::: "memory");
#endif
}

#if defined(HAVE_TDM)
// Issue one TDM 2D tile load: BK rows x HD f32 columns, row stride DIM floats.
// D# packing per CDNA5 ISA ch.8 (group0: count/lds/global/type, group1: dims).
__device__ __forceinline__ void tdm_load_tile(unsigned int lds_off, const float* gsrc) {
    unsigned long long ga = (unsigned long long)(uintptr_t)gsrc;
    v4u g0;
    g0[0] = 1u;                                   // count = 1 valid descriptor
    g0[1] = lds_off;                              // lds_addr (bytes)
    g0[2] = (unsigned int)ga;                     // global_addr[31:0]
    g0[3] = (unsigned int)((ga >> 32) & 0x1FFFFFFu) | (2u << 30); // addr[56:32] | type=2
    v8i g1;
    g1[0] = (int)(2u << 16);                      // data_size = 2 (4 bytes)
    g1[1] = (int)(((unsigned)HD & 0xFFFFu) << 16);// tensor_dim0 low16 = 128
    g1[2] = (int)(((unsigned)BK & 0xFFFFu) << 16);// tensor_dim1 low16 = 32
    g1[3] = (int)(((unsigned)HD) << 16);          // tile_dim0 = 128
    g1[4] = (int)BK;                              // tile_dim1 = 32, tile_dim2 = 0
    g1[5] = (int)DIM;                             // tensor_dim0_stride = 1024 (low32)
    g1[6] = 0;
    g1[7] = 0;
    v4i z4 = {0, 0, 0, 0};
#if defined(TDM_6ARG)
    v8i z8 = {0, 0, 0, 0, 0, 0, 0, 0};
    __builtin_amdgcn_tensor_load_to_lds(g0, g1, z4, z4, z8, 0);
#else
    __builtin_amdgcn_tensor_load_to_lds(g0, g1, z4, z4, 0);
#endif
}
#endif

// ---------------------------------------------------------------------------
// Flash attention: one workgroup = (b, h, 128-query tile); one wave = 16 rows.
// TDM prefetches the next K/V f32 tile into LDS while WMMAs run on the
// current f16 tile; a cooperative LDS->LDS pass converts f32 -> f16.
// ---------------------------------------------------------------------------
__global__ __launch_bounds__(256) void attn_fa_kernel(
        const float* __restrict__ Q, const float* __restrict__ K,
        const float* __restrict__ V, float* __restrict__ O) {
    __shared__ float    stage_k[BK * HD];     // TDM destination (f32) 16KB
    __shared__ float    stage_v[BK * HD];     // TDM destination (f32) 16KB
    __shared__ _Float16 lds_k[BK * HD];       // [key][dim]  f16
    __shared__ _Float16 lds_vt[HD * BK];      // [dim][key]  f16 (transposed)
    __shared__ _Float16 p_stage[8][16 * BK];  // per-wave P re-layout buffer

    const int tid  = threadIdx.x;
    const int wave = tid >> 5;
    const int lane = tid & 31;
    const int col  = lane & 15;
    const int hi   = lane >> 4;

    const int blk   = blockIdx.x;        // 0..255
    const int qtile = blk & 15;          // NLEN/BQ = 16
    const int h     = (blk >> 4) & 7;
    const int b     = blk >> 7;

    const int   qbase = qtile * BQ + wave * 16;
    const float scale = 0.08838834764831845f; // 1/sqrt(HD)

    // --- Load Q tile into four 16x32 f16 A-fragments (scale folded in) ---
    const float* qrow = Q + ((size_t)b * NLEN + qbase + col) * DIM + h * HD;
    v16h aQ[4];
#pragma unroll
    for (int kc = 0; kc < 4; ++kc) {
#pragma unroll
        for (int g = 0; g < 8; ++g) {
            const int kk = hi * 16 + 2 * g;
            float2 f = *(const float2*)(qrow + kc * 32 + kk);
            aQ[kc][2 * g]     = (_Float16)(f.x * scale);
            aQ[kc][2 * g + 1] = (_Float16)(f.y * scale);
        }
    }

    float rowm[8], rowl[8];
    v8f   o_acc[8];
#pragma unroll
    for (int r = 0; r < 8; ++r) { rowm[r] = -1e30f; rowl[r] = 0.0f; }
#pragma unroll
    for (int n = 0; n < 8; ++n) o_acc[n] = (v8f){0, 0, 0, 0, 0, 0, 0, 0};

    const size_t kv_base = ((size_t)b * SLEN) * DIM + h * HD;
    const float* kbase = K + kv_base;
    const float* vbase = V + kv_base;

#if defined(HAVE_TDM)
    const unsigned int off_sk = (unsigned int)(uintptr_t)(void*)stage_k;
    const unsigned int off_sv = (unsigned int)(uintptr_t)(void*)stage_v;
#endif

    // cooperative convert slot for this thread
    const int ckey = tid >> 3;        // 0..31
    const int cd0  = (tid & 7) * 16;  // 0..112

#if defined(HAVE_TDM)
    // ---- Prologue: TDM fetch block 0, convert, then prefetch block 1 ----
    if (wave == 0) {
        tdm_load_tile(off_sk, kbase + (size_t)0 * BK * DIM);
        tdm_load_tile(off_sv, vbase + (size_t)0 * BK * DIM);
        wait_tensorcnt0();
    }
    __syncthreads();
#pragma unroll
    for (int t = 0; t < 4; ++t) {
        float4 kf = *(const float4*)&stage_k[ckey * HD + cd0 + 4 * t];
        h4 kh = { (_Float16)kf.x, (_Float16)kf.y, (_Float16)kf.z, (_Float16)kf.w };
        *(h4*)&lds_k[ckey * HD + cd0 + 4 * t] = kh;
        float4 vf = *(const float4*)&stage_v[ckey * HD + cd0 + 4 * t];
        lds_vt[(cd0 + 4 * t + 0) * BK + ckey] = (_Float16)vf.x;
        lds_vt[(cd0 + 4 * t + 1) * BK + ckey] = (_Float16)vf.y;
        lds_vt[(cd0 + 4 * t + 2) * BK + ckey] = (_Float16)vf.z;
        lds_vt[(cd0 + 4 * t + 3) * BK + ckey] = (_Float16)vf.w;
    }
    __syncthreads();
    if (wave == 0 && NBLK > 1) {
        tdm_load_tile(off_sk, kbase + (size_t)1 * BK * DIM);
        tdm_load_tile(off_sv, vbase + (size_t)1 * BK * DIM);
    }
#endif

    for (int sblk = 0; sblk < NBLK; ++sblk) {
#if !defined(HAVE_TDM)
        // Fallback: direct global staging (f32 -> f16 convert in registers)
        {
            const float* kp = kbase + (size_t)(sblk * BK + ckey) * DIM + cd0;
            const float* vp = vbase + (size_t)(sblk * BK + ckey) * DIM + cd0;
#pragma unroll
            for (int t = 0; t < 4; ++t) {
                float4 kf = *(const float4*)(kp + 4 * t);
                h4 kh = { (_Float16)kf.x, (_Float16)kf.y,
                          (_Float16)kf.z, (_Float16)kf.w };
                *(h4*)&lds_k[ckey * HD + cd0 + 4 * t] = kh;
                float4 vf = *(const float4*)(vp + 4 * t);
                lds_vt[(cd0 + 4 * t + 0) * BK + ckey] = (_Float16)vf.x;
                lds_vt[(cd0 + 4 * t + 1) * BK + ckey] = (_Float16)vf.y;
                lds_vt[(cd0 + 4 * t + 2) * BK + ckey] = (_Float16)vf.z;
                lds_vt[(cd0 + 4 * t + 3) * BK + ckey] = (_Float16)vf.w;
            }
        }
        __syncthreads();
#endif

        // --- Scores: S = Q*K^T for 16x32 keys (two 16x16 tiles) ---
        v8f s0 = (v8f){0, 0, 0, 0, 0, 0, 0, 0};
        v8f s1 = (v8f){0, 0, 0, 0, 0, 0, 0, 0};
#pragma unroll
        for (int kc = 0; kc < 4; ++kc) {
            v16h b0 = *(const v16h*)&lds_k[(col)      * HD + kc * 32 + hi * 16];
            v16h b1 = *(const v16h*)&lds_k[(16 + col) * HD + kc * 32 + hi * 16];
            s0 = wmma_f16(aQ[kc], b0, s0);
            s1 = wmma_f16(aQ[kc], b1, s1);
        }

        // --- Online softmax over the 32 new columns ---
#pragma unroll
        for (int r = 0; r < 8; ++r) {
            float a0 = s0[r], a1 = s1[r];
            float bm = fmaxf(a0, a1);
#pragma unroll
            for (int m = 8; m >= 1; m >>= 1)
                bm = fmaxf(bm, __shfl_xor(bm, m, 32));
            const float nm = fmaxf(rowm[r], bm);
            const float p0 = __expf(a0 - nm);
            const float p1 = __expf(a1 - nm);
            float bs = p0 + p1;
#pragma unroll
            for (int m = 8; m >= 1; m >>= 1)
                bs += __shfl_xor(bs, m, 32);
            const float corr = __expf(rowm[r] - nm);
            rowl[r] = rowl[r] * corr + bs;
            rowm[r] = nm;
#pragma unroll
            for (int n = 0; n < 8; ++n) o_acc[n][r] *= corr;
            p_stage[wave][(r + 8 * hi) * BK + col]      = (_Float16)p0;
            p_stage[wave][(r + 8 * hi) * BK + 16 + col] = (_Float16)p1;
        }

        // --- Reload P as a 16x32 f16 A-fragment (same-wave LDS, in-order) ---
        v16h aP;
#pragma unroll
        for (int g = 0; g < 8; ++g) {
            const int kk = hi * 16 + 2 * g;
            h2 p = *(const h2*)&p_stage[wave][col * BK + kk];
            aP[2 * g]     = p.x;
            aP[2 * g + 1] = p.y;
        }

        // --- O += P * V  (8 output-dim chunks of 16) ---
#pragma unroll
        for (int n = 0; n < 8; ++n) {
            v16h bV = *(const v16h*)&lds_vt[(n * 16 + col) * BK + hi * 16];
            o_acc[n] = wmma_f16(aP, bV, o_acc[n]);
        }

#if defined(HAVE_TDM)
        if (sblk + 1 < NBLK) {
            __syncthreads();                 // all waves done with f16 buffers
            if (wave == 0) wait_tensorcnt0();// TDM for block sblk+1 landed
            __syncthreads();
#pragma unroll
            for (int t = 0; t < 4; ++t) {    // convert stage f32 -> f16 buffers
                float4 kf = *(const float4*)&stage_k[ckey * HD + cd0 + 4 * t];
                h4 kh = { (_Float16)kf.x, (_Float16)kf.y,
                          (_Float16)kf.z, (_Float16)kf.w };
                *(h4*)&lds_k[ckey * HD + cd0 + 4 * t] = kh;
                float4 vf = *(const float4*)&stage_v[ckey * HD + cd0 + 4 * t];
                lds_vt[(cd0 + 4 * t + 0) * BK + ckey] = (_Float16)vf.x;
                lds_vt[(cd0 + 4 * t + 1) * BK + ckey] = (_Float16)vf.y;
                lds_vt[(cd0 + 4 * t + 2) * BK + ckey] = (_Float16)vf.z;
                lds_vt[(cd0 + 4 * t + 3) * BK + ckey] = (_Float16)vf.w;
            }
            __syncthreads();                 // stage buffers free again
            if (wave == 0 && sblk + 2 < NBLK) {
                tdm_load_tile(off_sk, kbase + (size_t)(sblk + 2) * BK * DIM);
                tdm_load_tile(off_sv, vbase + (size_t)(sblk + 2) * BK * DIM);
            }
        }
#else
        __syncthreads();
#endif
    }

    // --- Normalize and store (f32 intermediate in workspace) ---
#pragma unroll
    for (int r = 0; r < 8; ++r) {
        const int   row = qbase + r + 8 * hi;
        const float inv = 1.0f / rowl[r];
        float* op = O + ((size_t)b * NLEN + row) * DIM + h * HD + col;
#pragma unroll
        for (int n = 0; n < 8; ++n)
            op[n * 16] = o_acc[n][r] * inv;
    }
}

// ---------------------------------------------------------------------------
// Projection: Y[4096,1024] = X @ W^T + bias, one 16x16 tile per wave.
// ---------------------------------------------------------------------------
__global__ __launch_bounds__(256) void proj_gemm_kernel(
        const float* __restrict__ X, const float* __restrict__ W,
        const float* __restrict__ bias, float* __restrict__ Y) {
    const int tid  = threadIdx.x;
    const int wave = tid >> 5;
    const int lane = tid & 31;
    const int col  = lane & 15;
    const int hi   = lane >> 4;

    const int tile  = blockIdx.x * 8 + wave;  // 16384 tiles total
    const int mtile = tile >> 6;              // 256 M-tiles
    const int ntile = tile & 63;              // 64  N-tiles

    const float* xrow = X + (size_t)(mtile * 16 + col) * DIM;
    const float* wrow = W + (size_t)(ntile * 16 + col) * DIM; // W[n][k]

    const float bv = bias[ntile * 16 + col];
    v8f c;
#pragma unroll
    for (int r = 0; r < 8; ++r) c[r] = bv;

    for (int kc = 0; kc < DIM / 32; ++kc) {
        v16h aX;
#pragma unroll
        for (int g = 0; g < 8; ++g) {
            const int kk = hi * 16 + 2 * g;
            float2 f = *(const float2*)(xrow + kc * 32 + kk);
            aX[2 * g]     = (_Float16)f.x;
            aX[2 * g + 1] = (_Float16)f.y;
        }
        v16h bW;
        const float* wp = wrow + kc * 32 + hi * 16;
#pragma unroll
        for (int j = 0; j < 16; j += 4) {
            float4 f = *(const float4*)(wp + j);
            bW[j]     = (_Float16)f.x;
            bW[j + 1] = (_Float16)f.y;
            bW[j + 2] = (_Float16)f.z;
            bW[j + 3] = (_Float16)f.w;
        }
        c = wmma_f16(aX, bW, c);
    }

#pragma unroll
    for (int r = 0; r < 8; ++r) {
        Y[(size_t)(mtile * 16 + r + 8 * hi) * DIM + ntile * 16 + col] = c[r];
    }
}

extern "C" void kernel_launch(void* const* d_in, const int* in_sizes, int n_in,
                              void* d_out, int out_size, void* d_ws, size_t ws_size,
                              hipStream_t stream) {
    const float* q  = (const float*)d_in[0];
    const float* k  = (const float*)d_in[1];
    const float* v  = (const float*)d_in[2];
    const float* pw = (const float*)d_in[3];
    const float* pb = (const float*)d_in[4];
    float* out = (float*)d_out;
    float* ws  = (float*)d_ws;   // 4096*1024 f32 attention output (16 MB)

    attn_fa_kernel<<<dim3(BATCH * NH * (NLEN / BQ)), dim3(256), 0, stream>>>(q, k, v, ws);
    proj_gemm_kernel<<<dim3(2048), dim3(256), 0, stream>>>(ws, pw, pb, out);
}
